// GINNet_7052336300584
// MI455X (gfx1250) — matrix operand order repaired and checked by reading, not compiled, not run on hardware
//
#include <hip/hip_runtime.h>
#include <hip/hip_bf16.h>

#define N_NODES 100000
#define N_EDGES 1600000
#define IN_DIM  32
#define HID     128
#define N_TILES (N_NODES / 16)   // 6250 exactly

typedef _Float16 v16h __attribute__((ext_vector_type(16)));
typedef float    v8f  __attribute__((ext_vector_type(8)));
typedef unsigned int u32;

// ---------------------------------------------------------------------------
// Kernel 1: agg[i] = (1 + eps) * x[i]
// ---------------------------------------------------------------------------
__global__ void gin_init(const float* __restrict__ x,
                         const float* __restrict__ eps,
                         float* __restrict__ agg) {
    int i = blockIdx.x * blockDim.x + threadIdx.x;
    if (i < N_NODES * IN_DIM) {
        agg[i] = (1.0f + eps[0]) * x[i];
    }
}

// ---------------------------------------------------------------------------
// Kernel 2: edge scatter, 8 threads per edge (4 floats each, float4 gather)
// ---------------------------------------------------------------------------
__global__ void gin_scatter(const float* __restrict__ x,
                            const long long* __restrict__ edge_index,
                            float* __restrict__ agg) {
    int idx = blockIdx.x * blockDim.x + threadIdx.x;
    if (idx >= N_EDGES * 8) return;
    int e = idx >> 3;
    int c = idx & 7;
    int src = (int)edge_index[e];
    int dst = (int)edge_index[N_EDGES + e];
    float4 v = *(const float4*)(x + src * IN_DIM + c * 4);
    float* d = agg + dst * IN_DIM + c * 4;
    atomicAdd(d + 0, v.x);
    atomicAdd(d + 1, v.y);
    atomicAdd(d + 2, v.z);
    atomicAdd(d + 3, v.w);
}

// ---------------------------------------------------------------------------
// Kernel 3: WMMA MLP chain. Transposed formulation: D[outfeat][node] = Wt x H.
// ---------------------------------------------------------------------------
__device__ __forceinline__ v8f wmma16(v16h a, v16h b, v8f c) {
    return __builtin_amdgcn_wmma_f32_16x16x32_f16(
        false, a, false, b, (short)0, c, false, false);
}

// SWAPX16 lane exchange: ds_swizzle group-of-32, xor=0x10, and=0x1f
__device__ __forceinline__ u32 swap16_u32(u32 x) {
    return (u32)__builtin_amdgcn_ds_swizzle((int)x, 0x401F);
}
__device__ __forceinline__ float swap16_f32(float x) {
    return __int_as_float(__builtin_amdgcn_ds_swizzle(__float_as_int(x), 0x401F));
}

// Convert two adjacent f32 C-tiles (feats [base,base+16), [base+16,base+32))
// into the f16 B operand of the next 32-wide K chunk. Pack to f16x2 dwords
// first, then exchange only 4 dwords between lane-halves.
__device__ __forceinline__ v16h make_b(v8f t0, v8f t1, int bh) {
    union HU { _Float16 h[2]; u32 u; };
    u32 p0[4], p1[4];
#pragma unroll
    for (int i = 0; i < 4; ++i) {
        HU a; a.h[0] = (_Float16)t0[2 * i]; a.h[1] = (_Float16)t0[2 * i + 1];
        HU b; b.h[0] = (_Float16)t1[2 * i]; b.h[1] = (_Float16)t1[2 * i + 1];
        p0[i] = a.u; p1[i] = b.u;
    }
    union OV { u32 u[8]; v16h v; } out;
#pragma unroll
    for (int i = 0; i < 4; ++i) {
        u32 send = bh ? p0[i] : p1[i];
        u32 recv = swap16_u32(send);
        out.u[i]     = bh ? recv : p0[i];   // feats (base+bh*16)+0..7
        out.u[4 + i] = bh ? p1[i] : recv;   // feats (base+bh*16)+8..15
    }
    return out.v;
}

#define NFRAG 56

__global__ void __launch_bounds__(256)
gin_mlp(const float* __restrict__ agg,
        const float* __restrict__ W1a, const float* __restrict__ b1a,
        const float* __restrict__ W1b, const float* __restrict__ b1b,
        const float* __restrict__ W2a, const float* __restrict__ b2a,
        const float* __restrict__ W2b, const float* __restrict__ b2b,
        const float* __restrict__ W2c, const float* __restrict__ b2c,
        float* __restrict__ out) {
    // Pre-swizzled A fragments: frag f, lane l -> 16 f16 at [f*512 + l*16]
    __shared__ __align__(32) _Float16 s_frag[NFRAG * 512];
    __shared__ __align__(32) float s_b1a[HID], s_b1b[IN_DIM];
    __shared__ __align__(32) float s_b2a[HID], s_b2b[HID], s_w2c[HID];

    // ---- cooperative weight swizzle: A[m][k] = W[k][m], contiguous k order, k = 16*(l>>4) + j
    for (int w = threadIdx.x; w < NFRAG * 32; w += blockDim.x) {
        int f = w >> 5, l = w & 31;
        int ml = l & 15, kh = l >> 4;
        const float* W; int Nout, mbase, kbase;
        if (f < 8)       { W = W1a; Nout = HID;    mbase = f * 16;               kbase = 0; }
        else if (f < 16) { W = W1b; Nout = IN_DIM; mbase = ((f - 8) >> 2) * 16;  kbase = ((f - 8) & 3) * 32; }
        else if (f < 24) { W = W2a; Nout = HID;    mbase = (f - 16) * 16;        kbase = 0; }
        else             { W = W2b; Nout = HID;    mbase = ((f - 24) >> 2) * 16; kbase = ((f - 24) & 3) * 32; }
        _Float16* dst = s_frag + f * 512 + l * 16;
#pragma unroll
        for (int j = 0; j < 16; ++j) {
            int kl = j + kh * 16;
            dst[j] = (_Float16)W[(kbase + kl) * Nout + mbase + ml];
        }
    }
    for (int i = threadIdx.x; i < HID; i += blockDim.x) {
        s_b1a[i] = b1a[i]; s_b2a[i] = b2a[i]; s_b2b[i] = b2b[i]; s_w2c[i] = W2c[i];
    }
    for (int i = threadIdx.x; i < IN_DIM; i += blockDim.x) s_b1b[i] = b1b[i];
    __syncthreads();

    const float bias_out = b2c[0];
    int lane = threadIdx.x & 31;
    int n    = lane & 15;      // node within tile (B/C column)
    int bh   = lane >> 4;      // lane-half
    int waveId = (blockIdx.x * blockDim.x + threadIdx.x) >> 5;
    int nWaves = (gridDim.x * blockDim.x) >> 5;

    for (int tile = waveId; tile < N_TILES; tile += nWaves) {
        int node = tile * 16 + n;

        // Opaque zero, re-created every iteration: all LDS reads below index
        // through it, so they stay ds_load_b128 (shared provenance kept) but
        // cannot be hoisted out of the loop -> no 400-VGPR live set, no spills.
        unsigned z = 0;
        asm volatile("" : "+v"(z));
        const _Float16* fb = s_frag + z + lane * 16;   // frag base, imm offsets
        const float* biasz = s_b1a;                    // arrays indexed with +z

        // ---- input B operand: feats bh*16 .. bh*16+15 of node n (f32 -> f16)
        const float* hrow = agg + node * IN_DIM + bh * 16;
        float4 q0 = *(const float4*)(hrow + 0);
        float4 q1 = *(const float4*)(hrow + 4);
        float4 q2 = *(const float4*)(hrow + 8);
        float4 q3 = *(const float4*)(hrow + 12);
        float fin[16] = {q0.x, q0.y, q0.z, q0.w, q1.x, q1.y, q1.z, q1.w,
                         q2.x, q2.y, q2.z, q2.w, q3.x, q3.y, q3.z, q3.w};
        v16h bin;
#pragma unroll
        for (int j = 0; j < 16; ++j) bin[j] = (_Float16)fin[j];

        // ---- layer 1: 32 -> 128, relu
        v8f acc1[8];
#pragma unroll
        for (int t = 0; t < 8; ++t) {
            v8f zc = {};
            v16h a = *(const v16h*)(fb + t * 512);
            acc1[t] = wmma16(a, bin, zc);
        }
#pragma unroll
        for (int t = 0; t < 8; ++t) {
            v8f bp = *(const v8f*)(s_b1a + z + t * 16 + bh * 8);
#pragma unroll
            for (int r = 0; r < 8; ++r) acc1[t][r] = fmaxf(acc1[t][r] + bp[r], 0.0f);
        }

        // ---- layer 2: 128 -> 32 (no relu; end of inner MLP)
        v8f acc2[2] = {{}, {}};
#pragma unroll
        for (int kc = 0; kc < 4; ++kc) {
            v16h b = make_b(acc1[2 * kc], acc1[2 * kc + 1], bh);
#pragma unroll
            for (int t = 0; t < 2; ++t) {
                v16h a = *(const v16h*)(fb + (8 + t * 4 + kc) * 512);
                acc2[t] = wmma16(a, b, acc2[t]);
            }
        }
#pragma unroll
        for (int t = 0; t < 2; ++t) {
            v8f bp = *(const v8f*)(s_b1b + z + t * 16 + bh * 8);
#pragma unroll
            for (int r = 0; r < 8; ++r) acc2[t][r] += bp[r];
        }

        // ---- layer 3: 32 -> 128, relu
        v16h b3 = make_b(acc2[0], acc2[1], bh);
        v8f acc3[8];
#pragma unroll
        for (int t = 0; t < 8; ++t) {
            v8f zc = {};
            v16h a = *(const v16h*)(fb + (16 + t) * 512);
            acc3[t] = wmma16(a, b3, zc);
        }
#pragma unroll
        for (int t = 0; t < 8; ++t) {
            v8f bp = *(const v8f*)(s_b2a + z + t * 16 + bh * 8);
#pragma unroll
            for (int r = 0; r < 8; ++r) acc3[t][r] = fmaxf(acc3[t][r] + bp[r], 0.0f);
        }

        // ---- layer 4: 128 -> 128, relu fused into final dot below
        v8f acc4[8] = {{}, {}, {}, {}, {}, {}, {}, {}};
#pragma unroll
        for (int kc = 0; kc < 4; ++kc) {
            v16h b = make_b(acc3[2 * kc], acc3[2 * kc + 1], bh);
#pragma unroll
            for (int t = 0; t < 8; ++t) {
                v16h a = *(const v16h*)(fb + (24 + t * 4 + kc) * 512);
                acc4[t] = wmma16(a, b, acc4[t]);
            }
        }

        // ---- layer 5: bias+relu + 128 -> 1 dot, sigmoid
        float partial = 0.0f;
#pragma unroll
        for (int t = 0; t < 8; ++t) {
            v8f bp = *(const v8f*)(s_b2b + z + t * 16 + bh * 8);
            v8f wp = *(const v8f*)(s_w2c + z + t * 16 + bh * 8);
#pragma unroll
            for (int r = 0; r < 8; ++r)
                partial += fmaxf(acc4[t][r] + bp[r], 0.0f) * wp[r];
        }
        (void)biasz;
        float total = partial + swap16_f32(partial);
        if (bh == 0) {
            out[node] = 1.0f / (1.0f + __expf(-(total + bias_out)));
        }
    }
}

// ---------------------------------------------------------------------------
extern "C" void kernel_launch(void* const* d_in, const int* in_sizes, int n_in,
                              void* d_out, int out_size, void* d_ws, size_t ws_size,
                              hipStream_t stream) {
    const float*     x   = (const float*)d_in[0];
    const long long* ei  = (const long long*)d_in[1];
    const float*     eps = (const float*)d_in[2];
    const float* W1a = (const float*)d_in[3];
    const float* b1a = (const float*)d_in[4];
    const float* W1b = (const float*)d_in[5];
    const float* b1b = (const float*)d_in[6];
    const float* W2a = (const float*)d_in[7];
    const float* b2a = (const float*)d_in[8];
    const float* W2b = (const float*)d_in[9];
    const float* b2b = (const float*)d_in[10];
    const float* W2c = (const float*)d_in[11];
    const float* b2c = (const float*)d_in[12];
    float* agg = (float*)d_ws;                  // N_NODES*32 f32 = 12.8 MB
    float* out = (float*)d_out;

    gin_init<<<(N_NODES * IN_DIM) / 256, 256, 0, stream>>>(x, eps, agg);
    gin_scatter<<<(N_EDGES * 8) / 256, 256, 0, stream>>>(x, ei, agg);
    gin_mlp<<<128, 256, 0, stream>>>(agg, W1a, b1a, W1b, b1b,
                                     W2a, b2a, W2b, b2b, W2c, b2c, out);
}
